// HolographicMemory_6777458393449
// MI455X (gfx1250) — matrix operand order repaired and checked by reading, not compiled, not run on hardware
//
#include <hip/hip_runtime.h>
#include <hip/hip_bf16.h>

typedef __attribute__((ext_vector_type(16))) _Float16 v16h;
typedef __attribute__((ext_vector_type(8)))  _Float16 v8h;
typedef __attribute__((ext_vector_type(8)))  float    v8f;

#define MEM_D   512
#define FLATD   4096
#define NBATCH  32768
#define DECAYF  0.95f
#define MTILES  2          // 16-row M-tiles per wave (B-fragment reuse factor)

// ---------------------------------------------------------------------------
// Kernel 1: p = pos_vec / ||pos_vec||   (1 block, 512 threads)
// ---------------------------------------------------------------------------
__global__ __launch_bounds__(512) void prep_p_kernel(const float* __restrict__ pos,
                                                     float* __restrict__ p) {
    __shared__ float red[16];
    __shared__ float tot;
    int t = threadIdx.x;                 // 0..511
    float v  = pos[t];
    float ss = v * v;
#pragma unroll
    for (int off = 16; off > 0; off >>= 1)
        ss += __shfl_xor(ss, off, 32);
    if ((t & 31) == 0) red[t >> 5] = ss;
    __syncthreads();
    if (t == 0) {
        float s = 0.f;
#pragma unroll
        for (int i = 0; i < 16; ++i) s += red[i];
        tot = s;
    }
    __syncthreads();
    p[t] = v * rsqrtf(tot);
}

// ---------------------------------------------------------------------------
// Kernel 2: pre-pack circulant M[k][n] = p[(k-n) mod 512] as f16 WMMA
// B-fragments, fragment-major:  Bf[((kt*32+nt)*32 + lane)*16 + j]
// lane -> n = nt*16 + (lane&15); k = kt*32 + (lane<16?0:8) + j + (j<8?0:8)
// Grid: 512 blocks (one fragment each) x 32 threads.
// ---------------------------------------------------------------------------
__global__ __launch_bounds__(32) void prep_B_kernel(const float* __restrict__ p,
                                                    _Float16* __restrict__ Bf) {
    int frag = blockIdx.x;               // 0..511
    int kt   = frag >> 5;                // 0..15
    int nt   = frag & 31;                // 0..31
    int lane = threadIdx.x;              // 0..31
    int n  = nt * 16 + (lane & 15);
    int kb = kt * 32 + ((lane & 16) ? 8 : 0);
    v16h vv;
#pragma unroll
    for (int j = 0; j < 16; ++j)
        vv[j] = (_Float16)p[(kb + j + (j < 8 ? 0 : 8) - n) & (MEM_D - 1)];
    *(v16h*)(Bf + ((size_t)frag * 32 + lane) * 16) = vv;   // 32B/lane, coalesced
}

// Load one 16x32 f16 A-fragment from a per-wave LDS row panel.
__device__ __forceinline__ v16h load_a_frag(const _Float16* A, int kt, int lane) {
    int m  = lane & 15;
    int hi = (lane & 16) ? 8 : 0;
    const _Float16* base = A + m * MEM_D + kt * 32 + hi;
    v8h lo = *(const v8h*)(base);        // K = kt*32 + hi + [0..7]
    v8h hh = *(const v8h*)(base + 16);   // K = kt*32 + 16 + hi + [0..7]
    return __builtin_shufflevector(lo, hh, 0,1,2,3,4,5,6,7,8,9,10,11,12,13,14,15);
}

// ---------------------------------------------------------------------------
// Main fused kernel: compress+normalize -> WMMA GEMM vs circulant -> decay add
// -> row L2 normalize.  One wave per block; each wave owns 2 M-tiles
// (32 batch rows) so every B fragment load feeds TWO wmma ops.
// ---------------------------------------------------------------------------
__global__ __launch_bounds__(32) void holo_main_kernel(
        const float* __restrict__ act, const float* __restrict__ prev,
        const _Float16* __restrict__ Bf, float* __restrict__ out) {
    __shared__ _Float16 sA[MTILES][16][MEM_D];   // 32 KB
    __shared__ float    snorm[MTILES][16];

    int lane = threadIdx.x & 31;
    int row0 = blockIdx.x * (16 * MTILES);       // first of 32 rows for this wave

    // ---- Phase 1: compressed = mean-of-8, row L2 normalize, stash f16 in LDS
    for (int mt = 0; mt < MTILES; ++mt) {
        for (int r = 0; r < 16; ++r) {
            const float* src = act + (size_t)(row0 + mt * 16 + r) * FLATD;
            float cv[16];
            float ss = 0.f;
#pragma unroll
            for (int seg = 0; seg < 16; ++seg) {
                int d = seg * 32 + lane;                 // lanes -> coalesced 32B
                const float4* q = (const float4*)(src + d * 8);
                float4 x0 = q[0], x1 = q[1];
                float s = x0.x + x0.y + x0.z + x0.w + x1.x + x1.y + x1.z + x1.w;
                float c = s * 0.125f;
                cv[seg] = c;
                ss += c * c;
            }
#pragma unroll
            for (int off = 16; off > 0; off >>= 1)
                ss += __shfl_xor(ss, off, 32);
            float rn = rsqrtf(ss);
#pragma unroll
            for (int seg = 0; seg < 16; ++seg)
                sA[mt][r][seg * 32 + lane] = (_Float16)(cv[seg] * rn);
        }
    }
    // Single-wave workgroup: DS ops from this wave are in-order; no barrier.

    // ---- Phase 2: GEMM (32 rows x 512 cols, K=512) with v_wmma_f32_16x16x32_f16
    float rowsq[MTILES][8];
#pragma unroll
    for (int mt = 0; mt < MTILES; ++mt)
#pragma unroll
        for (int v = 0; v < 8; ++v) rowsq[mt][v] = 0.f;

    int rb = (lane & 16) ? 8 : 0;       // accumulator row base for this lane half
    int cl = lane & 15;                 // accumulator column within tile
    const v16h* Bfrag = (const v16h*)Bf;

    for (int chunk = 0; chunk < 4; ++chunk) {            // 4 x 128 output columns
        v8f acc[MTILES][8] = {};
        for (int kt = 0; kt < 16; ++kt) {                // K = 512 / 32
            v16h a0 = load_a_frag(&sA[0][0][0], kt, lane);
            v16h a1 = load_a_frag(&sA[1][0][0], kt, lane);
#pragma unroll
            for (int t = 0; t < 8; ++t) {
                int nt = chunk * 8 + t;
                v16h b = Bfrag[(size_t)(kt * 32 + nt) * 32 + lane];
                acc[0][t] = __builtin_amdgcn_wmma_f32_16x16x32_f16(
                    false, a0, false, b, (short)0, acc[0][t], false, false);
                acc[1][t] = __builtin_amdgcn_wmma_f32_16x16x32_f16(
                    false, a1, false, b, (short)0, acc[1][t], false, false);
            }
        }
        // epilogue: mem = 0.95*prev + bound ; write unnormalized, gather sum(m^2)
#pragma unroll
        for (int mt = 0; mt < MTILES; ++mt) {
#pragma unroll
            for (int t = 0; t < 8; ++t) {
                int col = (chunk * 8 + t) * 16 + cl;
#pragma unroll
                for (int v = 0; v < 8; ++v) {
                    size_t idx = (size_t)(row0 + mt * 16 + rb + v) * MEM_D + col;
                    float m = DECAYF * prev[idx] + acc[mt][t][v];
                    out[idx] = m;
                    rowsq[mt][v] += m * m;
                }
            }
        }
    }

    // ---- Phase 3: finish row norms, rescale out in place (re-reads hit L2)
#pragma unroll
    for (int mt = 0; mt < MTILES; ++mt) {
#pragma unroll
        for (int v = 0; v < 8; ++v) {
            float s = rowsq[mt][v];
            s += __shfl_xor(s, 1, 32);
            s += __shfl_xor(s, 2, 32);
            s += __shfl_xor(s, 4, 32);
            s += __shfl_xor(s, 8, 32);  // full sum across the 16 columns
            if (cl == 0) snorm[mt][rb + v] = rsqrtf(s);
        }
    }
    asm volatile("s_wait_storecnt 0" ::: "memory");   // our stores visible to our loads

    for (int mt = 0; mt < MTILES; ++mt) {
        for (int r = 0; r < 16; ++r) {
            float rn = snorm[mt][r];
            size_t base = (size_t)(row0 + mt * 16 + r) * MEM_D;
#pragma unroll
            for (int seg = 0; seg < 4; ++seg) {
                float4* pq = (float4*)(out + base + seg * 128 + lane * 4);
                float4 m = *pq;
                m.x *= rn; m.y *= rn; m.z *= rn; m.w *= rn;
                *pq = m;
            }
        }
    }
}

// ---------------------------------------------------------------------------
extern "C" void kernel_launch(void* const* d_in, const int* in_sizes, int n_in,
                              void* d_out, int out_size, void* d_ws, size_t ws_size,
                              hipStream_t stream) {
    const float* act  = (const float*)d_in[0];   // (32768, 4096) f32
    const float* pos  = (const float*)d_in[1];   // (512,)        f32
    const float* prev = (const float*)d_in[2];   // (32768, 512)  f32
    float* out = (float*)d_out;                  // (32768, 512)  f32

    // workspace layout: [0, 512KB) packed B fragments (f16); then p (512 f32)
    _Float16* Bf = (_Float16*)d_ws;
    float*    p  = (float*)((char*)d_ws + (size_t)MEM_D * MEM_D * sizeof(_Float16));

    prep_p_kernel<<<1, 512, 0, stream>>>(pos, p);
    prep_B_kernel<<<512, 32, 0, stream>>>(p, Bf);

    int blocks = NBATCH / (16 * MTILES);         // 1024 single-wave blocks
    holo_main_kernel<<<blocks, 32, 0, stream>>>(act, prev, Bf, out);
}
